// Alpha_LSTMLayer_69372311765452
// MI455X (gfx1250) — compile-verified, matrix operand order checked by reading, not run on hardware
//
#include <hip/hip_runtime.h>
#include <hip/hip_bf16.h>

// ---------------------------------------------------------------------------
// Alpha-LSTM layer for MI455X (gfx1250).
//   B=64, T=512, D=512, H=1024.
// Design: single persistent kernel runs the whole T=512 recurrence with a
// grid-wide barrier per step. bf16 WMMA (v_wmma_f32_16x16x32_bf16) with fp32
// accumulation. Working set (~46MB) is L2-resident. Wave0 pipelines the input
// projection one step ahead of wave1's recurrent GEMM + cell update.
// ---------------------------------------------------------------------------

namespace {
constexpr int B = 64, T = 512, D = 512, H = 1024;
constexpr int NBLK = 256;                       // 4 batch-tiles * 64 h-col tiles
constexpr size_t HT_OFF = (size_t)B * T * H;    // hT offset in d_out
constexpr size_t CT_OFF = HT_OFF + (size_t)B * H;
}

typedef __bf16 bf16_t;
typedef __attribute__((ext_vector_type(16))) __bf16 v16bf;
typedef __attribute__((ext_vector_type(8)))  __bf16 v8bf;
typedef __attribute__((ext_vector_type(8)))  float  v8f;

// Per-lane WMMA fragment load (A tile row-major MxK, or B^T tile N-major/K-contig).
// ISA 16-bit A layout: lanes 0-15 -> K chunks {k0..k0+7, k0+16..k0+23},
//                      lanes 16-31 -> {k0+8..k0+15, k0+24..k0+31}.
__device__ __forceinline__ v16bf load_frag(const bf16_t* __restrict__ rowp, int kh) {
  v8bf c0 = *(const v8bf*)(rowp + (kh << 3));
  v8bf c1 = *(const v8bf*)(rowp + 16 + (kh << 3));
  return __builtin_shufflevector(c0, c1, 0,1,2,3,4,5,6,7,8,9,10,11,12,13,14,15);
}

__device__ __forceinline__ v8f wmma_bf16(v16bf a, v16bf b, v8f c) {
  return __builtin_amdgcn_wmma_f32_16x16x32_bf16(false, a, false, b, (short)0, c,
                                                 false, false);
}

__device__ __forceinline__ float sigf(float x) { return 1.0f / (1.0f + __expf(-x)); }
__device__ __forceinline__ float tanhfast(float x) { return 2.0f * sigf(2.0f * x) - 1.0f; }

// Grid-wide barrier: count + monotone generation counter in scratch.
__device__ __forceinline__ void grid_sync(int* cnt, int* gen, int target) {
  __syncthreads();
  if (threadIdx.x == 0) {
    __threadfence();
    if (atomicAdd(cnt, 1) == NBLK - 1) {
      atomicExch(cnt, 0);
      __threadfence();
      atomicAdd(gen, 1);
    }
    while (__hip_atomic_load(gen, __ATOMIC_ACQUIRE, __HIP_MEMORY_SCOPE_AGENT) < target) {
      __builtin_amdgcn_s_sleep(2);
    }
    __threadfence();
  }
  __syncthreads();
}

__global__ void cvt_f32_bf16(const float* __restrict__ s, bf16_t* __restrict__ d, int n) {
  int i = blockIdx.x * blockDim.x + threadIdx.x;
  int stride = gridDim.x * blockDim.x;
  for (; i < n; i += stride) d[i] = (bf16_t)s[i];
}

__global__ void init_state(const float* __restrict__ h0, const float* __restrict__ c0,
                           float* __restrict__ hbuf, float* __restrict__ cbuf,
                           bf16_t* __restrict__ hbfA, int* __restrict__ syncp, int n) {
  int i = blockIdx.x * blockDim.x + threadIdx.x;
  if (i < n) {
    hbuf[i] = h0[i];
    cbuf[i] = c0[i];
    hbfA[i] = (bf16_t)h0[i];
  }
  if (i == 0) { syncp[0] = 0; syncp[1] = 0; }
}

__global__ __launch_bounds__(64) void lstm_persistent(
    const bf16_t* __restrict__ xbf,   // [B,T,D] bf16
    const bf16_t* __restrict__ wih,   // [4H,D]  bf16 (N-major, K contig)
    const bf16_t* __restrict__ whh,   // [4H,H]  bf16 (N-major, K contig)
    const float*  __restrict__ bih,
    const float*  __restrict__ bhh,
    const float*  __restrict__ alphap,
    bf16_t* __restrict__ hbfA,        // bf16 h double-buffer (buf 0)
    bf16_t* __restrict__ hbfB,        // bf16 h double-buffer (buf 1)
    float*  __restrict__ hbuf,        // fp32 h carry
    float*  __restrict__ cbuf,        // fp32 c carry
    float*  __restrict__ dout,        // outputs[B,T,H] ++ hT ++ cT
    int*    __restrict__ syncp) {
  __shared__ float xp_lds[2][4][16][16];  // [pipe-buf][gate][m_local][n]

  const int wave  = threadIdx.x >> 5;
  const int lane  = threadIdx.x & 31;
  const int r16   = lane & 15;   // M (A) / N (B^T) row within tile
  const int kh    = lane >> 4;   // K-half selector
  const int mtile = blockIdx.x >> 6;   // batch tile 0..3
  const int jt    = blockIdx.x & 63;   // h-column tile 0..63
  const int brow  = mtile * 16 + r16;  // batch index for this lane's A rows
  const int hc    = jt * 16;
  const float alpha = *alphap;

  const bf16_t* wihr[4];
  const bf16_t* whhr[4];
  float bias[4];
#pragma unroll
  for (int g = 0; g < 4; ++g) {
    int grow = g * H + hc + r16;               // gate row (N index)
    wihr[g] = wih + (size_t)grow * D;
    whhr[g] = whh + (size_t)grow * H;
    bias[g] = bih[grow] + bhh[grow];
  }
  const bf16_t* xrow = xbf + (size_t)brow * T * D;

  // wave0: input projection for timestep tt -> xp_lds[tt&1]
  auto compute_xp = [&](int tt) {
    v8f a0 = {0,0,0,0,0,0,0,0}, a1 = a0, a2 = a0, a3 = a0;
    const bf16_t* ar = xrow + (size_t)tt * D;
#pragma unroll 4
    for (int k = 0; k < D; k += 32) {
      v16bf af = load_frag(ar + k, kh);
      a0 = wmma_bf16(af, load_frag(wihr[0] + k, kh), a0);
      a1 = wmma_bf16(af, load_frag(wihr[1] + k, kh), a1);
      a2 = wmma_bf16(af, load_frag(wihr[2] + k, kh), a2);
      a3 = wmma_bf16(af, load_frag(wihr[3] + k, kh), a3);
    }
    int buf = tt & 1;
#pragma unroll
    for (int r = 0; r < 8; ++r) {
      int ml = r + 8 * kh;                      // C/D layout: M = r + 8*half
      xp_lds[buf][0][ml][r16] = a0[r] + bias[0];
      xp_lds[buf][1][ml][r16] = a1[r] + bias[1];
      xp_lds[buf][2][ml][r16] = a2[r] + bias[2];
      xp_lds[buf][3][ml][r16] = a3[r] + bias[3];
    }
  };

  // wave1: recurrent GEMM + LSTM cell for timestep t
  auto compute_step = [&](int t, const bf16_t* hin, bf16_t* hout) {
    v8f ai = {0,0,0,0,0,0,0,0}, af_ = ai, ag_ = ai, ao_ = ai;
    const bf16_t* ar = hin + (size_t)brow * H;
#pragma unroll 4
    for (int k = 0; k < H; k += 32) {
      v16bf a = load_frag(ar + k, kh);
      ai  = wmma_bf16(a, load_frag(whhr[0] + k, kh), ai);
      af_ = wmma_bf16(a, load_frag(whhr[1] + k, kh), af_);
      ag_ = wmma_bf16(a, load_frag(whhr[2] + k, kh), ag_);
      ao_ = wmma_bf16(a, load_frag(whhr[3] + k, kh), ao_);
    }
    int buf = t & 1;
#pragma unroll
    for (int r = 0; r < 8; ++r) {
      int ml  = r + 8 * kh;
      int b   = mtile * 16 + ml;
      int col = hc + r16;
      size_t idx = (size_t)b * H + col;
      float xi = ai[r]  + xp_lds[buf][0][ml][r16];
      float xf = af_[r] + xp_lds[buf][1][ml][r16];
      float xg = ag_[r] + xp_lds[buf][2][ml][r16];
      float xo = ao_[r] + xp_lds[buf][3][ml][r16];
      float ig = sigf(xi), fg = sigf(xf), gg = tanhfast(xg), og = sigf(xo);
      float c_old = cbuf[idx], h_old = hbuf[idx];
      float c_new = fg * c_old + ig * gg;
      float h_new = og * tanhfast(c_new);
      float c_nx = alpha * c_new + (1.0f - alpha) * c_old;
      float h_nx = alpha * h_new + (1.0f - alpha) * h_old;
      cbuf[idx] = c_nx;
      hbuf[idx] = h_nx;
      hout[idx] = (bf16_t)h_nx;
      dout[((size_t)b * T + t) * H + col] = h_nx;
      if (t == T - 1) {
        dout[HT_OFF + idx] = h_nx;
        dout[CT_OFF + idx] = c_nx;
      }
    }
  };

  if (wave == 0) compute_xp(0);
  __syncthreads();

  for (int t = 0; t < T; ++t) {
    if (wave == 0) {
      if (t + 1 < T) compute_xp(t + 1);        // pipeline one step ahead
    } else {
      const bf16_t* hin = (t & 1) ? hbfB : hbfA;
      bf16_t*      hout = (t & 1) ? hbfA : hbfB;
      compute_step(t, hin, hout);
    }
    grid_sync(syncp, syncp + 1, t + 1);
  }
}

extern "C" void kernel_launch(void* const* d_in, const int* in_sizes, int n_in,
                              void* d_out, int out_size, void* d_ws, size_t ws_size,
                              hipStream_t stream) {
  const float* x     = (const float*)d_in[0];
  const float* h0    = (const float*)d_in[1];
  const float* c0    = (const float*)d_in[2];
  const float* W_ih  = (const float*)d_in[3];
  const float* W_hh  = (const float*)d_in[4];
  const float* b_ih  = (const float*)d_in[5];
  const float* b_hh  = (const float*)d_in[6];
  const float* alpha = (const float*)d_in[7];
  float* out = (float*)d_out;

  char* ws = (char*)d_ws;
  size_t off = 0;
  auto carve = [&](size_t bytes) -> char* {
    char* p = ws + off;
    off += (bytes + 255) & ~(size_t)255;
    return p;
  };
  bf16_t* xbf  = (bf16_t*)carve((size_t)B * T * D * 2);  // 32 MB
  bf16_t* wihb = (bf16_t*)carve((size_t)4 * H * D * 2);  //  4 MB
  bf16_t* whhb = (bf16_t*)carve((size_t)4 * H * H * 2);  //  8 MB
  bf16_t* hbfA = (bf16_t*)carve((size_t)B * H * 2);
  bf16_t* hbfB = (bf16_t*)carve((size_t)B * H * 2);
  float*  hbuf = (float*)carve((size_t)B * H * 4);
  float*  cbuf = (float*)carve((size_t)B * H * 4);
  int*    sync = (int*)carve(256);

  cvt_f32_bf16<<<2048, 256, 0, stream>>>(x, xbf, B * T * D);
  cvt_f32_bf16<<<512, 256, 0, stream>>>(W_ih, wihb, 4 * H * D);
  cvt_f32_bf16<<<512, 256, 0, stream>>>(W_hh, whhb, 4 * H * H);
  init_state<<<(B * H + 255) / 256, 256, 0, stream>>>(h0, c0, hbuf, cbuf, hbfA, sync,
                                                      B * H);
  lstm_persistent<<<NBLK, 64, 0, stream>>>(xbf, wihb, whhb, b_ih, b_hh, alpha, hbfA,
                                           hbfB, hbuf, cbuf, out, sync);
}
